// TextDecoder_44719199486597
// MI455X (gfx1250) — matrix operand order repaired and checked by reading, not compiled, not run on hardware
//
#include <hip/hip_runtime.h>
#include <hip/hip_bf16.h>

// ---------------------------------------------------------------------------
// Problem constants (from reference): B=64, T=32 (31 steps), F=1024, V=H=4000
// ---------------------------------------------------------------------------
#define BB     64
#define TT     32
#define FDIM   1024
#define HH     4000
#define G4     16000          // 4*H
#define LDW0   5024           // W_ih0 row stride (F + V)
#define KSPLIT 2016           // layer0 K split point (multiple of 32)

typedef __attribute__((ext_vector_type(16))) __bf16 v16bf;
typedef __attribute__((ext_vector_type(8)))  float  v8f;

union FragBF { v16bf v; uint4 q[2]; };

__device__ __forceinline__ v8f wmma_bf16(v16bf a, v16bf b, v8f c) {
    return __builtin_amdgcn_wmma_f32_16x16x32_bf16(false, a, false, b,
                                                   (short)0, c, false, false);
}

// A fragment (16x32 bf16, MxK). lane 0-15: M=lane, K={kk..kk+7, kk+16..kk+23};
// lane 16-31: M=lane-16, K shifted by 8. Two 16B loads per lane.
__device__ __forceinline__ v16bf ldA(const __hip_bfloat16* __restrict__ A,
                                     int lda, int row, int kk, int lh) {
    FragBF f;
    const uint4* p = reinterpret_cast<const uint4*>(A + (size_t)row * lda + kk + lh * 8);
    f.q[0] = p[0];
    f.q[1] = p[2];   // +16 halves
    return f.v;
}

// B fragment (32x16 bf16, KxN) taken from row-major W (B[k,n] = W[row_n, k]).
// lane 0-15: N=lane, K={kk..kk+7, kk+16..kk+23}; lane 16-31: N=lane-16, K shifted by 8.
__device__ __forceinline__ v16bf ldB(const __hip_bfloat16* __restrict__ W,
                                     int ldw, int row, int kk, int lh) {
    FragBF f;
    const uint4* p = reinterpret_cast<const uint4*>(W + (size_t)row * ldw + kk + lh * 8);
    f.q[0] = p[0];
    f.q[1] = p[2];
    return f.v;
}

__device__ __forceinline__ float sigm(float x) { return 1.0f / (1.0f + __expf(-x)); }

// Software-pipelined 64x16 (4 M-tiles) GEMM over K in [kbeg,kend).
// Unrolled by 2 so the double buffers rotate structurally (no register copies):
//   load set1 @ kk   -> WMMA set0 @ kk-32
//   load set0 @ kk+32 -> WMMA set1 @ kk
__device__ __forceinline__ void gemm_loop(const __hip_bfloat16* __restrict__ A, int lda,
                                          const __hip_bfloat16* __restrict__ W, int ldw,
                                          int wrow, int kbeg, int kend,
                                          int laneN, int lh, v8f acc[4]) {
    const __hip_bfloat16* wp = W + (size_t)wrow * ldw;
    const int n = (kend - kbeg) >> 5;           // number of K chunks
    v16bf b0 = ldB(W, ldw, wrow, kbeg, lh);
    v16bf a0[4];
    #pragma unroll
    for (int m = 0; m < 4; ++m) a0[m] = ldA(A, lda, m * 16 + laneN, kbeg, lh);
    v16bf b1;
    v16bf a1[4];
    int kk = kbeg + 32;
    #pragma unroll 1
    for (int i = 0; i < ((n - 1) >> 1); ++i, kk += 64) {
        __builtin_prefetch((const void*)(wp + kk + 2048), 0, 1);
        b1 = ldB(W, ldw, wrow, kk, lh);
        #pragma unroll
        for (int m = 0; m < 4; ++m) a1[m] = ldA(A, lda, m * 16 + laneN, kk, lh);
        #pragma unroll
        for (int m = 0; m < 4; ++m) acc[m] = wmma_bf16(a0[m], b0, acc[m]);
        b0 = ldB(W, ldw, wrow, kk + 32, lh);
        #pragma unroll
        for (int m = 0; m < 4; ++m) a0[m] = ldA(A, lda, m * 16 + laneN, kk + 32, lh);
        #pragma unroll
        for (int m = 0; m < 4; ++m) acc[m] = wmma_bf16(a1[m], b1, acc[m]);
    }
    if (n & 1) {
        #pragma unroll
        for (int m = 0; m < 4; ++m) acc[m] = wmma_bf16(a0[m], b0, acc[m]);
    } else {
        b1 = ldB(W, ldw, wrow, kk, lh);
        #pragma unroll
        for (int m = 0; m < 4; ++m) a1[m] = ldA(A, lda, m * 16 + laneN, kk, lh);
        #pragma unroll
        for (int m = 0; m < 4; ++m) acc[m] = wmma_bf16(a0[m], b0, acc[m]);
        #pragma unroll
        for (int m = 0; m < 4; ++m) acc[m] = wmma_bf16(a1[m], b1, acc[m]);
    }
}

// ---------------------------------------------------------------------------
// fp32 -> bf16 dense convert (n multiple of 4)
// ---------------------------------------------------------------------------
__global__ void k_cvt(const float* __restrict__ src, __hip_bfloat16* __restrict__ dst,
                      long n) {
    long i = ((long)blockIdx.x * blockDim.x + threadIdx.x) * 4;
    if (i >= n) return;
    float4 f = *reinterpret_cast<const float4*>(src + i);
    union { __hip_bfloat16 h[4]; uint2 u; } t;
    t.h[0] = __float2bfloat16(f.x); t.h[1] = __float2bfloat16(f.y);
    t.h[2] = __float2bfloat16(f.z); t.h[3] = __float2bfloat16(f.w);
    *reinterpret_cast<uint2*>(dst + i) = t.u;
}

// W_f = W_ih0[:, :FDIM] (row stride LDW0) -> bf16 (16000 x 1024)
__global__ void k_cvt_wf(const float* __restrict__ src, __hip_bfloat16* __restrict__ dst) {
    int r = blockIdx.x;
    int c = threadIdx.x * 4;
    float4 f = *reinterpret_cast<const float4*>(src + (size_t)r * LDW0 + c);
    union { __hip_bfloat16 h[4]; uint2 u; } t;
    t.h[0] = __float2bfloat16(f.x); t.h[1] = __float2bfloat16(f.y);
    t.h[2] = __float2bfloat16(f.z); t.h[3] = __float2bfloat16(f.w);
    *reinterpret_cast<uint2*>(dst + (size_t)r * FDIM + c) = t.u;
}

// bsum = b_ih + b_hh (pre-summed biases, 16000 each)
__global__ void k_bsum(const float* __restrict__ a, const float* __restrict__ b,
                       float* __restrict__ o, int n) {
    int i = blockIdx.x * blockDim.x + threadIdx.x;
    if (i < n) o[i] = a[i] + b[i];
}

// Transpose W_v = W_ih0[:, FDIM:] (16000 x 4000, row stride LDW0)
// -> WvT bf16 (4000 x 16000) so embedding rows are contiguous.
__global__ void k_transpose(const float* __restrict__ src, __hip_bfloat16* __restrict__ dst) {
    __shared__ float tile[32][33];
    int j0 = blockIdx.x * 32;     // gate-dim index (row of W_v)
    int v0 = blockIdx.y * 32;     // vocab index
    int tx = threadIdx.x & 31, ty = threadIdx.x >> 5;
    #pragma unroll
    for (int yy = 0; yy < 32; yy += 8)
        tile[ty + yy][tx] = src[(size_t)(j0 + ty + yy) * LDW0 + FDIM + v0 + tx];
    __syncthreads();
    #pragma unroll
    for (int yy = 0; yy < 32; yy += 8)
        dst[(size_t)(v0 + ty + yy) * G4 + j0 + tx] =
            __float2bfloat16(tile[tx][ty + yy]);
}

// feat_pre = features @ W_f^T  (64 x 16000), bf16 WMMA, fp32 out.
__global__ void __launch_bounds__(128)
k_featpre(const __hip_bfloat16* __restrict__ fb,   // 64 x 1024 bf16
          const __hip_bfloat16* __restrict__ Wf,   // 16000 x 1024 bf16
          float* __restrict__ fp) {                // 64 x 16000 f32
    int lane = threadIdx.x & 31, w = threadIdx.x >> 5;
    int laneN = lane & 15, lh = lane >> 4;
    int col0 = blockIdx.x * 64 + w * 16;
    int wrow = col0 + laneN;
    v8f acc[4] = {};
    gemm_loop(fb, FDIM, Wf, FDIM, wrow, 0, FDIM, laneN, lh, acc);
    #pragma unroll
    for (int m = 0; m < 4; ++m)
        #pragma unroll
        for (int r = 0; r < 8; ++r)
            fp[(size_t)(m * 16 + r + lh * 8) * G4 + col0 + laneN] = acc[m][r];
}

// ---------------------------------------------------------------------------
// Layer 0 per-step. 256 threads = 8 waves: wave = (gate g, K-half ks).
// g = h0@W_hh0^T + feat_pre + WvT[tok] + bsum0; LSTM cell -> h0_out, c0.
// ---------------------------------------------------------------------------
__global__ void __launch_bounds__(256)
k_layer0(const __hip_bfloat16* __restrict__ hin,   // 64 x 4000 bf16
         __hip_bfloat16* __restrict__ hout,        // 64 x 4000 bf16
         float* __restrict__ c,                    // 64 x 4000 f32 (in-place)
         const __hip_bfloat16* __restrict__ Whh,   // 16000 x 4000 bf16
         const float* __restrict__ featpre,        // 64 x 16000 f32
         const __hip_bfloat16* __restrict__ WvT,   // 4000 x 16000 bf16
         const int* __restrict__ captions,         // 64 x 32
         const float* __restrict__ bsum,           // 16000 (b_ih0+b_hh0)
         int t) {
    __shared__ float smem[8][64][16];
    int lane = threadIdx.x & 31, wv = threadIdx.x >> 5;     // wv = 0..7
    int g = wv >> 1, ks = wv & 1;                           // gate, K-half
    int laneN = lane & 15, lh = lane >> 4;
    int n0 = blockIdx.x * 16;                               // h-col tile
    int wrow = g * HH + n0 + laneN;
    int kbeg = ks ? KSPLIT : 0;
    int kend = ks ? HH : KSPLIT;
    v8f acc[4] = {};
    gemm_loop(hin, HH, Whh, HH, wrow, kbeg, kend, laneN, lh, acc);
    #pragma unroll
    for (int m = 0; m < 4; ++m)
        #pragma unroll
        for (int r = 0; r < 8; ++r)
            smem[wv][m * 16 + r + lh * 8][laneN] = acc[m][r];
    __syncthreads();
    #pragma unroll
    for (int i = 0; i < 4; ++i) {
        int idx = threadIdx.x + i * 256;
        int b_ = idx >> 4, n = idx & 15;
        int j  = n0 + n;
        int tok = captions[b_ * TT + t];
        const float* fpr = featpre + (size_t)b_ * G4;
        const __hip_bfloat16* er = WvT + (size_t)tok * G4;
        float gi = smem[0][b_][n] + smem[1][b_][n] + fpr[j]        + __bfloat162float(er[j])        + bsum[j];
        float gf = smem[2][b_][n] + smem[3][b_][n] + fpr[HH + j]   + __bfloat162float(er[HH + j])   + bsum[HH + j];
        float gg = smem[4][b_][n] + smem[5][b_][n] + fpr[2*HH + j] + __bfloat162float(er[2*HH + j]) + bsum[2*HH + j];
        float go = smem[6][b_][n] + smem[7][b_][n] + fpr[3*HH + j] + __bfloat162float(er[3*HH + j]) + bsum[3*HH + j];
        float cp = c[(size_t)b_ * HH + j];
        float cn = sigm(gf) * cp + sigm(gi) * tanhf(gg);
        float hn = sigm(go) * tanhf(cn);
        c[(size_t)b_ * HH + j]    = cn;
        hout[(size_t)b_ * HH + j] = __float2bfloat16(hn);
    }
}

// ---------------------------------------------------------------------------
// Layer 1 per-step. 256 threads = 8 waves: wave = (gate g, src ks).
// ks=0: h0@W_ih1^T over full K; ks=1: h1@W_hh1^T over full K; summed in LDS.
// ---------------------------------------------------------------------------
__global__ void __launch_bounds__(256)
k_layer1(const __hip_bfloat16* __restrict__ h0,    // 64 x 4000 bf16 (layer0 out)
         const __hip_bfloat16* __restrict__ h1in,  // 64 x 4000 bf16
         __hip_bfloat16* __restrict__ h1out,
         float* __restrict__ c,                    // c1
         const __hip_bfloat16* __restrict__ Wih,   // 16000 x 4000 bf16
         const __hip_bfloat16* __restrict__ Whh,   // 16000 x 4000 bf16
         const float* __restrict__ bsum,           // 16000 (b_ih1+b_hh1)
         float* __restrict__ out,                  // 64 x 32 x 4000 f32
         int t) {
    __shared__ float smem[8][64][16];
    int lane = threadIdx.x & 31, wv = threadIdx.x >> 5;
    int g = wv >> 1, ks = wv & 1;
    int laneN = lane & 15, lh = lane >> 4;
    int n0 = blockIdx.x * 16;
    int wrow = g * HH + n0 + laneN;
    const __hip_bfloat16* A = ks ? h1in : h0;
    const __hip_bfloat16* W = ks ? Whh : Wih;
    v8f acc[4] = {};
    gemm_loop(A, HH, W, HH, wrow, 0, HH, laneN, lh, acc);
    #pragma unroll
    for (int m = 0; m < 4; ++m)
        #pragma unroll
        for (int r = 0; r < 8; ++r)
            smem[wv][m * 16 + r + lh * 8][laneN] = acc[m][r];
    __syncthreads();
    #pragma unroll
    for (int i = 0; i < 4; ++i) {
        int idx = threadIdx.x + i * 256;
        int b_ = idx >> 4, n = idx & 15;
        int j  = n0 + n;
        float gi = smem[0][b_][n] + smem[1][b_][n] + bsum[j];
        float gf = smem[2][b_][n] + smem[3][b_][n] + bsum[HH + j];
        float gg = smem[4][b_][n] + smem[5][b_][n] + bsum[2*HH + j];
        float go = smem[6][b_][n] + smem[7][b_][n] + bsum[3*HH + j];
        float cp = c[(size_t)b_ * HH + j];
        float cn = sigm(gf) * cp + sigm(gi) * tanhf(gg);
        float hn = sigm(go) * tanhf(cn);
        c[(size_t)b_ * HH + j]     = cn;
        h1out[(size_t)b_ * HH + j] = __float2bfloat16(hn);
        out[(size_t)b_ * (TT * HH) + (size_t)(t + 1) * HH + j] = hn;
    }
}

// ---------------------------------------------------------------------------
extern "C" void kernel_launch(void* const* d_in, const int* in_sizes, int n_in,
                              void* d_out, int out_size, void* d_ws, size_t ws_size,
                              hipStream_t stream) {
    const float* features = (const float*)d_in[0];
    const int*   captions = (const int*)  d_in[1];
    const float* W_ih0    = (const float*)d_in[2];
    const float* W_hh0    = (const float*)d_in[3];
    const float* b_ih0    = (const float*)d_in[4];
    const float* b_hh0    = (const float*)d_in[5];
    const float* W_ih1    = (const float*)d_in[6];
    const float* W_hh1    = (const float*)d_in[7];
    const float* b_ih1    = (const float*)d_in[8];
    const float* b_hh1    = (const float*)d_in[9];
    float* out = (float*)d_out;

    char* ws = (char*)d_ws;
    size_t off = 0;
    auto alloc = [&](size_t bytes) -> char* {
        char* p = ws + off;
        off += (bytes + 255) & ~(size_t)255;
        return p;
    };

    const size_t NW = (size_t)G4 * HH;   // 64M elems per recurrent weight
    __hip_bfloat16* Wb_hh0 = (__hip_bfloat16*)alloc(NW * 2);
    __hip_bfloat16* Wb_ih1 = (__hip_bfloat16*)alloc(NW * 2);
    __hip_bfloat16* Wb_hh1 = (__hip_bfloat16*)alloc(NW * 2);
    __hip_bfloat16* WvT    = (__hip_bfloat16*)alloc((size_t)HH * G4 * 2);   // 4000 x 16000
    __hip_bfloat16* Wb_f   = (__hip_bfloat16*)alloc((size_t)G4 * FDIM * 2); // 16000 x 1024
    __hip_bfloat16* featb  = (__hip_bfloat16*)alloc((size_t)BB * FDIM * 2);
    float*          featpre= (float*)alloc((size_t)BB * G4 * 4);
    float*          bsum0  = (float*)alloc((size_t)G4 * 4);
    float*          bsum1  = (float*)alloc((size_t)G4 * 4);
    __hip_bfloat16* h0a    = (__hip_bfloat16*)alloc((size_t)BB * HH * 2);
    __hip_bfloat16* h0b    = (__hip_bfloat16*)alloc((size_t)BB * HH * 2);
    __hip_bfloat16* h1a    = (__hip_bfloat16*)alloc((size_t)BB * HH * 2);
    __hip_bfloat16* h1b    = (__hip_bfloat16*)alloc((size_t)BB * HH * 2);
    float*          c0     = (float*)alloc((size_t)BB * HH * 4);
    float*          c1     = (float*)alloc((size_t)BB * HH * 4);
    (void)in_sizes; (void)n_in; (void)ws_size;

    // Zero output (covers out[:,0,:]) and initial recurrent state.
    hipMemsetAsync(d_out, 0, (size_t)out_size * 4, stream);
    hipMemsetAsync(h0a, 0, (size_t)BB * HH * 2, stream);
    hipMemsetAsync(h1a, 0, (size_t)BB * HH * 2, stream);
    hipMemsetAsync(c0,  0, (size_t)BB * HH * 4, stream);
    hipMemsetAsync(c1,  0, (size_t)BB * HH * 4, stream);

    // One-time (per launch) weight preparation.
    k_cvt<<<62500, 256, 0, stream>>>(W_hh0, Wb_hh0, (long)NW);
    k_cvt<<<62500, 256, 0, stream>>>(W_ih1, Wb_ih1, (long)NW);
    k_cvt<<<62500, 256, 0, stream>>>(W_hh1, Wb_hh1, (long)NW);
    k_cvt<<<64,    256, 0, stream>>>(features, featb, (long)(BB * FDIM));
    k_cvt_wf<<<G4, 256, 0, stream>>>(W_ih0, Wb_f);
    k_transpose<<<dim3(G4 / 32, HH / 32), 256, 0, stream>>>(W_ih0, WvT);
    k_bsum<<<(G4 + 255) / 256, 256, 0, stream>>>(b_ih0, b_hh0, bsum0, G4);
    k_bsum<<<(G4 + 255) / 256, 256, 0, stream>>>(b_ih1, b_hh1, bsum1, G4);

    // feat_pre = features @ W_f^T
    k_featpre<<<G4 / 64, 128, 0, stream>>>(featb, Wb_f, featpre);

    // 31 recurrent steps; stream ordering provides the h0 -> layer1 sync.
    for (int t = 0; t < TT - 1; ++t) {
        const __hip_bfloat16* h0i = (t & 1) ? h0b : h0a;
        __hip_bfloat16*       h0o = (t & 1) ? h0a : h0b;
        const __hip_bfloat16* h1i = (t & 1) ? h1b : h1a;
        __hip_bfloat16*       h1o = (t & 1) ? h1a : h1b;
        k_layer0<<<HH / 16, 256, 0, stream>>>(h0i, h0o, c0, Wb_hh0, featpre,
                                              WvT, captions, bsum0, t);
        k_layer1<<<HH / 16, 256, 0, stream>>>(h0o, h1i, h1o, c1, Wb_ih1, Wb_hh1,
                                              bsum1, out, t);
    }
}
